// BertAttentionHead_81973745812060
// MI455X (gfx1250) — matrix operand order repaired and checked by reading, not compiled, not run on hardware
//
#include <hip/hip_runtime.h>
#include <hip/hip_bf16.h>
#include <stdint.h>

// ---------------------------------------------------------------------------
// BERT attention head, CDNA5 (gfx1250) wave32 + WMMA bf16 + TDM staging.
//   B=8, S=2048, E=1024, H=64.
// Kernel 0: one-time prep — W{q,k,v} fp32 [E,64] -> Wt bf16 [3][64][E]
//           (column-major, k-contiguous) so projection B-fragments are two
//           b128 loads instead of 16 strided b32 loads + converts.
// Kernel 1: QKV projection with v_wmma_f32_16x16x32_bf16. Emits
//           Q*(1/sqrt(H)) and K as [B,S,64] bf16, V transposed [B,64,S].
// Kernel 2: causal flash attention; K/V chunks staged to double-buffered LDS
//           via the Tensor Data Mover (overlapped with compute), fragments
//           read from LDS as contiguous b128s, online softmax with 16-lane
//           shfl reductions, PV accumulated in fp32.
// The boolean mask input is ignored: the mask is causal by construction.
// ---------------------------------------------------------------------------

typedef __attribute__((ext_vector_type(16))) __bf16 v16bf;
typedef __attribute__((ext_vector_type(8)))  __bf16 bf16x8;
typedef __attribute__((ext_vector_type(8)))  float  v8f;
typedef __attribute__((ext_vector_type(4)))  unsigned u32x4;
typedef __attribute__((ext_vector_type(8)))  int      i32x8;
typedef __attribute__((ext_vector_type(4)))  int      i32x4;

#define ATTN_B 8
#define ATTN_S 2048
#define ATTN_E 1024
#define ATTN_H 64

// ---- WMMA helper -----------------------------------------------------------
__device__ __forceinline__ v8f wmma_bf16(v16bf a, v16bf b, v8f c) {
  // (neg_a, A, neg_b, B, c_mod, C, reuse_a, reuse_b)
  return __builtin_amdgcn_wmma_f32_16x16x32_bf16(false, a, false, b,
                                                 (short)0, c, false, false);
}

// 16 contiguous bf16 -> one fragment slice (two b128 loads).
__device__ __forceinline__ v16bf ld16(const __bf16* p) {
  bf16x8 a = *(const bf16x8*)p;
  bf16x8 b = *(const bf16x8*)(p + 8);
  v16bf r;
#pragma unroll
  for (int i = 0; i < 8; ++i) { r[i] = a[i]; r[i + 8] = b[i]; }
  return r;
}

// A-fragment for a 16x32 bf16 tile: lane holds K {hi*8+0..7, 16+hi*8+0..7}
// of its row. `rowp` points at (row, K=0) of this lane's row.
__device__ __forceinline__ v16bf ld_afrag(const __bf16* rowp, int hi) {
  bf16x8 a = *(const bf16x8*)(rowp + hi * 8);
  bf16x8 b = *(const bf16x8*)(rowp + 16 + hi * 8);
  v16bf r;
#pragma unroll
  for (int i = 0; i < 8; ++i) { r[i] = a[i]; r[i + 8] = b[i]; }
  return r;
}

// ---- Tensor Data Mover: 2D tile of 2-byte elements, global -> LDS ----------
// 6-arg builtin on this toolchain:
//   (uint32x4 g0, int32x8 g1, int32x4 g2, int32x4 g3, int32x8 extra, i32 cpol)
__device__ __forceinline__ void tdm_load_2d(unsigned lds_off, const void* gptr,
                                            unsigned tile_d0, unsigned tile_d1,
                                            unsigned stride0,
                                            unsigned tensor_d0,
                                            unsigned tensor_d1) {
  unsigned long long ga = (unsigned long long)(uintptr_t)gptr;
  u32x4 g0;
  g0.x = 1u;                                     // count=1, user descriptor
  g0.y = lds_off;                                // lds_addr (bytes)
  g0.z = (unsigned)ga;                           // global_addr[31:0]
  g0.w = (unsigned)((ga >> 32) & 0x01FFFFFFu) | 0x80000000u; // addr[56:32]|type=2
  i32x8 g1;
  g1[0] = (int)(1u << 16);                       // data_size=1 (2B), no mask
  g1[1] = (int)(tensor_d0 << 16);                // tensor_dim0[15:0] @63:48
  g1[2] = (int)((tensor_d0 >> 16) | (tensor_d1 << 16)); // dim0 hi | dim1 lo
  g1[3] = (int)((tensor_d1 >> 16) | (tile_d0 << 16));   // dim1 hi | tile_dim0
  g1[4] = (int)(tile_d1 & 0xFFFF);               // tile_dim1 | tile_dim2=0
  g1[5] = (int)stride0;                          // tensor_dim0_stride lo32
  g1[6] = 0;
  g1[7] = 0;
  i32x4 z4 = {0, 0, 0, 0};
  i32x8 z8 = {0, 0, 0, 0, 0, 0, 0, 0};
  __builtin_amdgcn_tensor_load_to_lds(g0, g1, z4, z4, z8, 0);
}

// ---------------------------------------------------------------------------
// Kernel 0: prep. Wt[m][col][k] = (bf16) W_m[k][col].  3*64*1024 elements.
// ---------------------------------------------------------------------------
__global__ __launch_bounds__(256) void prep_wt_kernel(
    const float* __restrict__ Wq, const float* __restrict__ Wk,
    const float* __restrict__ Wv, __bf16* __restrict__ Wt) {
  const int idx = blockIdx.x * 256 + threadIdx.x;   // 0 .. 3*64*1024-1
  const int m   = idx >> 16;                        // / (64*1024)
  const int col = (idx >> 10) & 63;
  const int k   = idx & 1023;
  const float* W = (m == 0) ? Wq : (m == 1) ? Wk : Wv;
  Wt[idx] = (__bf16)W[k * ATTN_H + col];
}

// ---------------------------------------------------------------------------
// Kernel 1: QKV projection. Each wave: one 16-row tile of x (over B*S rows),
// all 3 weight matrices, full E reduction, bf16 WMMA w/ fp32 accumulate.
// ---------------------------------------------------------------------------
__global__ __launch_bounds__(128) void qkv_proj_kernel(
    const float* __restrict__ x, const __bf16* __restrict__ Wt,
    const float* __restrict__ bq, const float* __restrict__ bk,
    const float* __restrict__ bv,
    __bf16* __restrict__ Qo, __bf16* __restrict__ Ko,
    __bf16* __restrict__ Vto) {
  const int lane = threadIdx.x & 31;
  const int wave = threadIdx.x >> 5;
  const int lo = lane & 15;
  const int hi = lane >> 4;
  const int tile = blockIdx.x * 4 + wave;   // 0..1023 over B*S/16
  const int row0 = tile * 16;

  const float* xrow = x + (size_t)(row0 + lo) * ATTN_E;
  const __bf16* Wtq = Wt;                              // [64][1024]
  const __bf16* Wtk = Wt + 64 * ATTN_E;
  const __bf16* Wtv = Wt + 128 * ATTN_E;

  v8f acc[3][4];
#pragma unroll
  for (int m = 0; m < 3; ++m)
#pragma unroll
    for (int nt = 0; nt < 4; ++nt)
#pragma unroll
      for (int r = 0; r < 8; ++r) acc[m][nt][r] = 0.0f;

  for (int k = 0; k < ATTN_E; k += 32) {
    // A fragment: this lane's row, K = hi*16 + 0..15.
    float4 a0 = *(const float4*)(xrow + k + hi * 16);
    float4 a1 = *(const float4*)(xrow + k + hi * 16 + 4);
    float4 c0 = *(const float4*)(xrow + k + hi * 16 + 8);
    float4 c1 = *(const float4*)(xrow + k + hi * 16 + 12);
    v16bf a;
    a[0] = (__bf16)a0.x;  a[1] = (__bf16)a0.y;  a[2]  = (__bf16)a0.z;  a[3]  = (__bf16)a0.w;
    a[4] = (__bf16)a1.x;  a[5] = (__bf16)a1.y;  a[6]  = (__bf16)a1.z;  a[7]  = (__bf16)a1.w;
    a[8] = (__bf16)c0.x;  a[9] = (__bf16)c0.y;  a[10] = (__bf16)c0.z;  a[11] = (__bf16)c0.w;
    a[12]= (__bf16)c1.x;  a[13]= (__bf16)c1.y;  a[14] = (__bf16)c1.z;  a[15] = (__bf16)c1.w;

    // B fragments: Wt[col][k + 16*hi + 0..15] -> contiguous b128 pairs.
    const int wtoff = lo * ATTN_E + k + hi * 16;   // col = nt*16+lo adds nt*16*E
#pragma unroll
    for (int nt = 0; nt < 4; ++nt) {
      const int o = nt * 16 * ATTN_E + wtoff;
      v16bf bfq = ld16(Wtq + o);
      acc[0][nt] = wmma_bf16(a, bfq, acc[0][nt]);
      v16bf bfk = ld16(Wtk + o);
      acc[1][nt] = wmma_bf16(a, bfk, acc[1][nt]);
      v16bf bfv = ld16(Wtv + o);
      acc[2][nt] = wmma_bf16(a, bfv, acc[2][nt]);
    }
  }

  // Epilogue: add bias; fold the 1/sqrt(H)=0.125 softmax scale into Q.
#pragma unroll
  for (int nt = 0; nt < 4; ++nt) {
    const int col = nt * 16 + lo;
    const float qb = bq[col], kbias = bk[col], vb = bv[col];
#pragma unroll
    for (int r = 0; r < 8; ++r) {
      const int row = row0 + r + 8 * hi;             // global row over B*S
      Qo[(size_t)row * ATTN_H + col] = (__bf16)((acc[0][nt][r] + qb) * 0.125f);
      Ko[(size_t)row * ATTN_H + col] = (__bf16)(acc[1][nt][r] + kbias);
      const int bb = row >> 11;                       // row / S
      const int ss = row & (ATTN_S - 1);              // row % S
      Vto[((size_t)bb * ATTN_H + col) * ATTN_S + ss] = (__bf16)(acc[2][nt][r] + vb);
    }
  }
}

// ---------------------------------------------------------------------------
// Kernel 2: causal flash attention. Block = 4 waves = 64 query rows of one
// batch. K chunk (32x64) and V^T chunk (64x32) staged into double-buffered
// LDS by wave 0 via TDM, overlapped with compute on the other buffer.
// ---------------------------------------------------------------------------
__global__ __launch_bounds__(128) void flash_attn_kernel(
    const __bf16* __restrict__ Q, const __bf16* __restrict__ K,
    const __bf16* __restrict__ Vt, float* __restrict__ out) {
  __shared__ alignas(16) __bf16 Kl[2][32 * ATTN_H];   // [key][dim]   2x4 KB
  __shared__ alignas(16) __bf16 Vl[2][ATTN_H * 32];   // [dim][key]   2x4 KB
  __shared__ alignas(16) __bf16 Pl[4][16 * 32];       // per-wave P   4 KB

  const int lane = threadIdx.x & 31;
  const int wave = threadIdx.x >> 5;
  const int lo = lane & 15;
  const int hi = lane >> 4;
  const int b  = blockIdx.x >> 5;       // 32 query tiles (of 64 rows) per batch
  const int qt = blockIdx.x & 31;
  const int qbase = qt * 64;
  const int wq = qbase + wave * 16;     // this wave's first query row

  // Q fragments (A-layout) straight from global: row = wq + lo.
  const __bf16* Qrow = Q + ((size_t)b * ATTN_S + wq + lo) * ATTN_H;
  const v16bf qf0 = ld16(Qrow + hi * 16);        // dims 0..31
  const v16bf qf1 = ld16(Qrow + 32 + hi * 16);   // dims 32..63

  v8f o[4];
#pragma unroll
  for (int nt = 0; nt < 4; ++nt)
#pragma unroll
    for (int r = 0; r < 8; ++r) o[nt][r] = 0.0f;

  float m[8], l[8];
#pragma unroll
  for (int r = 0; r < 8; ++r) { m[r] = -3.0e38f; l[r] = 0.0f; }

  const __bf16* Kbase = K + (size_t)b * ATTN_S * ATTN_H;
  const __bf16* Vbase = Vt + (size_t)b * ATTN_H * ATTN_S;
  unsigned kl_off[2], vl_off[2];
#pragma unroll
  for (int p = 0; p < 2; ++p) {
    kl_off[p] = (unsigned)(uintptr_t)&Kl[p][0];
    vl_off[p] = (unsigned)(uintptr_t)&Vl[p][0];
  }

  const int nchunks = (qbase + 64) >> 5;        // causal: keys <= qbase+63

  // Prologue: stage chunk 0 into buffer 0.
  if (wave == 0) {
    tdm_load_2d(kl_off[0], Kbase, 32 * ATTN_H, 1, 32 * ATTN_H, 32 * ATTN_H, 1);
    tdm_load_2d(vl_off[0], Vbase, 32, ATTN_H, ATTN_S, ATTN_S, ATTN_H);
  }

  for (int c = 0; c < nchunks; ++c) {
    const int p = c & 1;
    const int kb = c * 32;

    if (wave == 0) __builtin_amdgcn_s_wait_tensorcnt(0);  // chunk c landed
    __syncthreads();     // all waves: chunk c visible, buffer p^1 free
    if (wave == 0 && c + 1 < nchunks) {
      // Overlap: stage chunk c+1 into the other buffer during compute.
      const int nb = kb + 32;
      tdm_load_2d(kl_off[p ^ 1], Kbase + (size_t)nb * ATTN_H,
                  32 * ATTN_H, 1, 32 * ATTN_H, 32 * ATTN_H, 1);
      tdm_load_2d(vl_off[p ^ 1], Vbase + nb,
                  32, ATTN_H, ATTN_S, ATTN_S, ATTN_H);
    }

    // ---- S = Q K^T (two 16-key subtiles, K-dim 64 = 2 WMMAs each) ----
    v8f st[2];
#pragma unroll
    for (int sub = 0; sub < 2; ++sub) {
      const __bf16* krow = &Kl[p][(sub * 16 + lo) * ATTN_H + hi * 16];
      v16bf kf0 = ld16(krow);         // dims 16*hi + 0..15
      v16bf kf1 = ld16(krow + 32);    // dims 32 + 16*hi + 0..15
      v8f s;
#pragma unroll
      for (int r = 0; r < 8; ++r) s[r] = 0.0f;
      s = wmma_bf16(qf0, kf0, s);
      s = wmma_bf16(qf1, kf1, s);
      st[sub] = s;
    }

    // ---- causal mask + online softmax (rows striped over 16-lane halves) ---
    float p0[8], p1[8], alpha[8];
#pragma unroll
    for (int r = 0; r < 8; ++r) {
      const int qg = wq + r + 8 * hi;           // global query row
      float s0 = (kb + lo      <= qg) ? st[0][r] : -3.0e38f;
      float s1 = (kb + 16 + lo <= qg) ? st[1][r] : -3.0e38f;
      float mx = fmaxf(s0, s1);
      mx = fmaxf(mx, __shfl_xor(mx, 1, 32));
      mx = fmaxf(mx, __shfl_xor(mx, 2, 32));
      mx = fmaxf(mx, __shfl_xor(mx, 4, 32));
      mx = fmaxf(mx, __shfl_xor(mx, 8, 32));
      const float mn = fmaxf(m[r], mx);
      const float al = __expf(m[r] - mn);
      const float e0 = __expf(s0 - mn);
      const float e1 = __expf(s1 - mn);
      float rs = e0 + e1;
      rs += __shfl_xor(rs, 1, 32);
      rs += __shfl_xor(rs, 2, 32);
      rs += __shfl_xor(rs, 4, 32);
      rs += __shfl_xor(rs, 8, 32);
      l[r] = l[r] * al + rs;
      m[r] = mn;
      alpha[r] = al; p0[r] = e0; p1[r] = e1;
    }

    // Rescale running output by alpha (per row).
#pragma unroll
    for (int nt = 0; nt < 4; ++nt)
#pragma unroll
      for (int r = 0; r < 8; ++r) o[nt][r] *= alpha[r];

    // ---- C-layout P -> A-layout via per-wave LDS round trip ----
#pragma unroll
    for (int r = 0; r < 8; ++r) {
      const int M = r + 8 * hi;
      Pl[wave][M * 32 + lo]      = (__bf16)p0[r];
      Pl[wave][M * 32 + 16 + lo] = (__bf16)p1[r];
    }
    const v16bf pf = ld16(&Pl[wave][lo * 32 + hi * 16]);

    // ---- O += P V (4 dim-subtiles of 16) ----
#pragma unroll
    for (int nt = 0; nt < 4; ++nt) {
      v16bf vf = ld16(&Vl[p][(nt * 16 + lo) * 32 + hi * 16]);
      o[nt] = wmma_bf16(pf, vf, o[nt]);
    }
  }

  // ---- normalize and store fp32 output [B,S,H] ----
#pragma unroll
  for (int nt = 0; nt < 4; ++nt)
#pragma unroll
    for (int r = 0; r < 8; ++r) {
      const int qg = wq + r + 8 * hi;
      out[((size_t)b * ATTN_S + qg) * ATTN_H + nt * 16 + lo] =
          o[nt][r] / l[r];
    }
}

// ---------------------------------------------------------------------------
extern "C" void kernel_launch(void* const* d_in, const int* in_sizes, int n_in,
                              void* d_out, int out_size, void* d_ws,
                              size_t ws_size, hipStream_t stream) {
  const float* x  = (const float*)d_in[0];
  // d_in[1] is the boolean mask: ignored (causal mask applied analytically).
  const float* Wq = (const float*)d_in[2];
  const float* bq = (const float*)d_in[3];
  const float* Wk = (const float*)d_in[4];
  const float* bk = (const float*)d_in[5];
  const float* Wv = (const float*)d_in[6];
  const float* bv = (const float*)d_in[7];
  float* out = (float*)d_out;

  const size_t n_tok = (size_t)ATTN_B * ATTN_S;          // 16384 rows
  __bf16* Qbf = (__bf16*)d_ws;                           // [B*S, 64]
  __bf16* Kbf = Qbf + n_tok * ATTN_H;                    // [B*S, 64]
  __bf16* Vtb = Kbf + n_tok * ATTN_H;                    // [B, 64, S]
  __bf16* Wt  = Vtb + n_tok * ATTN_H;                    // [3, 64, 1024]

  // Kernel 0: weight transpose+convert, 3*64*1024 elements.
  prep_wt_kernel<<<768, 256, 0, stream>>>(Wq, Wk, Wv, Wt);
  // Kernel 1: 1024 row-tiles / 4 waves per block -> 256 blocks.
  qkv_proj_kernel<<<256, 128, 0, stream>>>(x, Wt, bq, bk, bv,
                                           Qbf, Kbf, Vtb);
  // Kernel 2: B * (S/64) = 256 blocks of 4 waves.
  flash_attn_kernel<<<256, 128, 0, stream>>>(Qbf, Kbf, Vtb, out);
}
